// _JointTypedGraphActionRanker_51823075393573
// MI455X (gfx1250) — compile-verified, matrix operand order checked by reading, not run on hardware
//
#include <hip/hip_runtime.h>
#include <hip/hip_bf16.h>

typedef __bf16 bf16_t;
typedef __attribute__((ext_vector_type(16))) __bf16 bf16x16;
typedef __attribute__((ext_vector_type(8)))  __bf16 bf16x8;
typedef __attribute__((ext_vector_type(8)))  float  f32x8;

#define H      64
#define NNODE  50000
#define NEDGE  800000
#define NACT   512
#define NFEAT  128
#define NPRIM  16
#define LN_EPS 1e-5f

// ---------------------------------------------------------------- WMMA core
__device__ __forceinline__ f32x8 wmma_bf16(bf16x16 a, bf16x16 b, f32x8 c) {
    return __builtin_amdgcn_wmma_f32_16x16x32_bf16(false, a, false, b,
                                                   (short)0, c, false, false);
}

// A-fragment (16x32 bf16, M=lane&15): lane<16 holds K = kbase+{0..7,16..23},
// lane>=16 holds K = kbase+8+{0..7,16..23}. khalf = (lane>=16)?8:0.
__device__ __forceinline__ bf16x16 afrag_bf16(const bf16_t* row, int kbase, int khalf) {
    bf16x8 lo = *reinterpret_cast<const bf16x8*>(row + kbase + khalf);
    bf16x8 hi = *reinterpret_cast<const bf16x8*>(row + kbase + khalf + 16);
    bf16x16 r;
#pragma unroll
    for (int e = 0; e < 8; ++e) { r[e] = lo[e]; r[e + 8] = hi[e]; }
    return r;
}

__device__ __forceinline__ bf16x16 afrag_f32(const float* row, int kbase, int khalf) {
    const float* p = row + kbase + khalf;
    bf16x16 r;
#pragma unroll
    for (int e = 0; e < 8; ++e) {
        r[e]     = (bf16_t)p[e];
        r[e + 8] = (bf16_t)p[e + 16];
    }
    return r;
}

// B fragments pre-packed: frag (chunk c, tile t) for lane l is 16 contiguous bf16.
__device__ __forceinline__ bf16x16 bfrag(const bf16_t* pack, int c, int t, int lane) {
    return reinterpret_cast<const bf16x16*>(pack)[(c * 4 + t) * 32 + lane];
}

// Store C-layout accumulators (bias+ReLU) as bf16 16xH tile into wave-private LDS.
__device__ __forceinline__ void store_hidden_lds(bf16_t* hb, const f32x8* acc,
                                                 int lane, const float* bias) {
    int nl = lane & 15, half = lane >> 4;
#pragma unroll
    for (int t = 0; t < 4; ++t) {
        float b = bias[t * 16 + nl];
#pragma unroll
        for (int r = 0; r < 8; ++r) {
            float v = acc[t][r] + b;
            v = v > 0.f ? v : 0.f;
            hb[(r + 8 * half) * H + t * 16 + nl] = (bf16_t)v;
        }
    }
    asm volatile("s_wait_dscnt 0" ::: "memory");  // cross-lane LDS RAW inside wave
}

// Second 64->64 GEMM layer reading A-frags out of the LDS hidden tile.
__device__ __forceinline__ void gemm2_64(f32x8* acc2, const bf16_t* hb,
                                         const bf16_t* wpack, int lane) {
    int khalf = (lane >= 16) ? 8 : 0;
    const bf16_t* row = hb + (lane & 15) * H;
#pragma unroll
    for (int c = 0; c < 2; ++c) {
        bf16x16 a = afrag_bf16(row, c * 32, khalf);
#pragma unroll
        for (int t = 0; t < 4; ++t)
            acc2[t] = wmma_bf16(a, bfrag(wpack, c, t, lane), acc2[t]);
    }
}

// ---------------------------------------------------------------- utilities
__global__ void k_pack_b(const float* __restrict__ W, int K, int kchunks,
                         bf16_t* __restrict__ out) {
    int idx = blockIdx.x * 256 + threadIdx.x;
    int total = kchunks * 4 * 32 * 16;
    if (idx >= total) return;
    int e = idx & 15, lane = (idx >> 4) & 31, ft = idx >> 9;
    int c = ft >> 2, t = ft & 3;
    int n = t * 16 + (lane & 15);
    int k = c * 32 + ((lane >= 16) ? 8 : 0) + (e < 8 ? e : 8 + e);
    float v = (k < K) ? W[k * H + n] : 0.f;
    out[idx] = (bf16_t)v;
}

__global__ void k_f32_to_bf16(const float* __restrict__ in, bf16_t* __restrict__ out, int n) {
    int i = blockIdx.x * 256 + threadIdx.x;
    if (i < n) out[i] = (bf16_t)in[i];
}

__global__ void k_zero_f32(float* __restrict__ p, int n) {
    int i = blockIdx.x * 256 + threadIdx.x;
    if (i < n) p[i] = 0.f;
}

// ---------------------------------------------------------------- node MLP
__global__ void k_node_mlp(const float* __restrict__ feat,
                           const bf16_t* __restrict__ wn1p, const float* __restrict__ bn1,
                           const bf16_t* __restrict__ wn2p, const float* __restrict__ bn2,
                           float* __restrict__ nodes_f32, bf16_t* __restrict__ nodes_bf16) {
    __shared__ bf16_t hb[8][16 * H];
    int lane = threadIdx.x & 31, w = threadIdx.x >> 5;
    int rowbase = (blockIdx.x * 8 + w) * 16;
    int myrow = rowbase + (lane & 15);
    int ldrow = myrow < NNODE ? myrow : NNODE - 1;
    int khalf = (lane >= 16) ? 8 : 0;
    const float* arow = feat + (long)ldrow * NFEAT;

    f32x8 acc[4] = {};
#pragma unroll
    for (int c = 0; c < 4; ++c) {
        bf16x16 a = afrag_f32(arow, c * 32, khalf);
#pragma unroll
        for (int t = 0; t < 4; ++t)
            acc[t] = wmma_bf16(a, bfrag(wn1p, c, t, lane), acc[t]);
    }
    store_hidden_lds(hb[w], acc, lane, bn1);

    f32x8 acc2[4] = {};
    gemm2_64(acc2, hb[w], wn2p, lane);

    int nl = lane & 15, half = lane >> 4;
#pragma unroll
    for (int t = 0; t < 4; ++t) {
        float b = bn2[t * 16 + nl];
#pragma unroll
        for (int r = 0; r < 8; ++r) {
            int m = rowbase + r + 8 * half;
            if (m < NNODE) {
                float v = acc2[t][r] + b;
                v = v > 0.f ? v : 0.f;
                nodes_f32[(long)m * H + t * 16 + nl]  = v;
                nodes_bf16[(long)m * H + t * 16 + nl] = (bf16_t)v;
            }
        }
    }
}

// ---------------------------------------------------------------- edge MLP + scatter
__global__ void k_edge_mlp(const bf16_t* __restrict__ nodes_bf16,
                           const bf16_t* __restrict__ rel_bf16,
                           const float* __restrict__ eweight,
                           const int* __restrict__ esrc, const int* __restrict__ edst,
                           const int* __restrict__ erel,
                           const bf16_t* __restrict__ wm1p, const float* __restrict__ bm1,
                           const bf16_t* __restrict__ wm2p, const float* __restrict__ bm2,
                           float* __restrict__ agg) {
    __shared__ bf16_t hb[8][16 * H];
    int lane = threadIdx.x & 31, w = threadIdx.x >> 5;
    long ebase = (long)(blockIdx.x * 8 + w) * 16;
    long e = ebase + (lane & 15);
    int src = esrc[e], dst = edst[e], rel = erel[e];
    float wgt = eweight[e];
    (void)dst;
    int khalf = (lane >= 16) ? 8 : 0;
    const bf16_t* srow = nodes_bf16 + (long)src * H;
    const bf16_t* drow = nodes_bf16 + (long)edst[e] * H;
    const bf16_t* rrow = rel_bf16 + rel * H;

    f32x8 acc[4] = {};
#pragma unroll
    for (int c = 0; c < 7; ++c) {
        bf16x16 a;
        if (c < 2)      a = afrag_bf16(srow, c * 32, khalf);
        else if (c < 4) a = afrag_bf16(drow, (c - 2) * 32, khalf);
        else if (c < 6) a = afrag_bf16(rrow, (c - 4) * 32, khalf);
        else {
#pragma unroll
            for (int ix = 0; ix < 16; ++ix) a[ix] = (bf16_t)0.f;
            if (khalf == 0) a[0] = (bf16_t)wgt;  // K = 192 : edge weight
        }
#pragma unroll
        for (int t = 0; t < 4; ++t)
            acc[t] = wmma_bf16(a, bfrag(wm1p, c, t, lane), acc[t]);
    }
    store_hidden_lds(hb[w], acc, lane, bm1);

    f32x8 acc2[4] = {};
    gemm2_64(acc2, hb[w], wm2p, lane);

    int nl = lane & 15, half = lane >> 4;
    int d2row[8];
#pragma unroll
    for (int r = 0; r < 8; ++r) d2row[r] = edst[ebase + r + 8 * half];
#pragma unroll
    for (int t = 0; t < 4; ++t) {
        float b = bm2[t * 16 + nl];
#pragma unroll
        for (int r = 0; r < 8; ++r)
            atomicAdd(agg + (long)d2row[r] * H + t * 16 + nl, acc2[t][r] + b);
    }
}

// ---------------------------------------------------------------- layernorm + graph mean
__global__ void k_layernorm(const float* __restrict__ nodes_f32, const float* __restrict__ agg,
                            const float* __restrict__ g, const float* __restrict__ b,
                            bf16_t* __restrict__ ln_bf16, float* __restrict__ gsum) {
    __shared__ float gacc[H];
    int tid = threadIdx.x;
    if (tid < H) gacc[tid] = 0.f;
    __syncthreads();
    int lane = tid & 31, w = tid >> 5;
    int row = blockIdx.x * 8 + w;
    if (row < NNODE) {
        int n0 = lane * 2;
        float x0 = nodes_f32[(long)row * H + n0]     + agg[(long)row * H + n0];
        float x1 = nodes_f32[(long)row * H + n0 + 1] + agg[(long)row * H + n0 + 1];
        float s = x0 + x1, sq = x0 * x0 + x1 * x1;
#pragma unroll
        for (int m = 1; m < 32; m <<= 1) {
            s  += __shfl_xor(s,  m, 32);
            sq += __shfl_xor(sq, m, 32);
        }
        float mu  = s * (1.f / H);
        float var = sq * (1.f / H) - mu * mu;
        float inv = rsqrtf(var + LN_EPS);
        float y0 = (x0 - mu) * inv * g[n0]     + b[n0];
        float y1 = (x1 - mu) * inv * g[n0 + 1] + b[n0 + 1];
        ln_bf16[(long)row * H + n0]     = (bf16_t)y0;
        ln_bf16[(long)row * H + n0 + 1] = (bf16_t)y1;
        atomicAdd(&gacc[n0], y0);
        atomicAdd(&gacc[n0 + 1], y1);
    }
    __syncthreads();
    if (tid < H) atomicAdd(&gsum[tid], gacc[tid]);
}

__global__ void k_graph_finalize(const float* __restrict__ gsum, bf16_t* __restrict__ graph_bf) {
    int t = threadIdx.x;
    if (t < H) graph_bf[t] = (bf16_t)(gsum[t] * (1.f / NNODE));
}

// ---------------------------------------------------------------- action encoder
__global__ void k_action_enc(const bf16_t* __restrict__ ln_bf16, const bf16_t* __restrict__ graph_bf,
                             const int* __restrict__ cblock, const int* __restrict__ ctarget,
                             const int* __restrict__ cprim, const int* __restrict__ htarget,
                             const bf16_t* __restrict__ wa1p, const float* __restrict__ ba1,
                             const bf16_t* __restrict__ wa2p, const float* __restrict__ ba2,
                             bf16_t* __restrict__ enc_bf) {
    __shared__ bf16_t hb[8][16 * H];
    int lane = threadIdx.x & 31, w = threadIdx.x >> 5;
    int base = (blockIdx.x * 8 + w) * 16;
    int a = base + (lane & 15);
    int cb = cblock[a], ct = ctarget[a], prim = cprim[a];
    float ht = (float)htarget[a];
    int khalf = (lane >= 16) ? 8 : 0;
    const bf16_t* brow = ln_bf16 + (long)cb * H;
    const bf16_t* trow = ln_bf16 + (long)ct * H;

    bf16x16 bfv[2], tfv[2], dfv[2];
#pragma unroll
    for (int c = 0; c < 2; ++c) {
        bfv[c] = afrag_bf16(brow, c * 32, khalf);
        bf16x16 t0 = afrag_bf16(trow, c * 32, khalf);
#pragma unroll
        for (int ix = 0; ix < 16; ++ix) {
            float tv = (float)t0[ix] * ht;
            t0[ix] = (bf16_t)tv;
            float dv = (float)bfv[c][ix] - tv;
            dfv[c][ix] = (bf16_t)(dv < 0.f ? -dv : dv);
        }
        tfv[c] = t0;
    }

    f32x8 acc[4] = {};
#pragma unroll
    for (int c = 0; c < 9; ++c) {
        bf16x16 afr;
        if (c < 2)      afr = bfv[c];
        else if (c < 4) afr = tfv[c - 2];
        else if (c < 6) afr = afrag_bf16(graph_bf, (c - 4) * 32, khalf);
        else if (c < 8) afr = dfv[c - 6];
        else {
#pragma unroll
            for (int ix = 0; ix < 16; ++ix) {
                int kk = khalf + (ix < 8 ? ix : 8 + ix);  // offset inside chunk 8
                float v = 0.f;
                if (kk < NPRIM)      v = (kk == prim) ? 1.f : 0.f;  // one-hot 256..271
                else if (kk == NPRIM) v = ht;                        // 272: has_target
                afr[ix] = (bf16_t)v;
            }
        }
#pragma unroll
        for (int t = 0; t < 4; ++t)
            acc[t] = wmma_bf16(afr, bfrag(wa1p, c, t, lane), acc[t]);
    }
    store_hidden_lds(hb[w], acc, lane, ba1);

    f32x8 acc2[4] = {};
    gemm2_64(acc2, hb[w], wa2p, lane);

    int nl = lane & 15, half = lane >> 4;
#pragma unroll
    for (int t = 0; t < 4; ++t) {
        float bb = ba2[t * 16 + nl];
#pragma unroll
        for (int r = 0; r < 8; ++r) {
            float v = acc2[t][r] + bb;
            v = v > 0.f ? v : 0.f;
            enc_bf[(long)(base + r + 8 * half) * H + t * 16 + nl] = (bf16_t)v;
        }
    }
}

// ---------------------------------------------------------------- pairwise ranker
__global__ void k_pair_rank(const bf16_t* __restrict__ enc_bf,
                            const bf16_t* __restrict__ wp1p, const float* __restrict__ bp1,
                            const float* __restrict__ wp2, const float* __restrict__ bp2,
                            float* __restrict__ out) {
    int lane = threadIdx.x & 31, w = threadIdx.x >> 5;
    int flat = blockIdx.x * 8 + w;   // 0 .. A*A/16 - 1
    int i  = flat >> 5;              // lhs index (A/16 groups of 16 j's)
    int j0 = (flat & 31) * 16;
    int khalf = (lane >= 16) ? 8 : 0;
    const bf16_t* lrow = enc_bf + (long)i * H;
    const bf16_t* rrow = enc_bf + (long)(j0 + (lane & 15)) * H;

    bf16x16 lf[2], rf[2], df[2], af[2];
#pragma unroll
    for (int c = 0; c < 2; ++c) {
        lf[c] = afrag_bf16(lrow, c * 32, khalf);
        rf[c] = afrag_bf16(rrow, c * 32, khalf);
#pragma unroll
        for (int ix = 0; ix < 16; ++ix) {
            float dv = (float)lf[c][ix] - (float)rf[c][ix];
            df[c][ix] = (bf16_t)dv;
            af[c][ix] = (bf16_t)(dv < 0.f ? -dv : dv);
        }
    }

    f32x8 acc[4] = {};
#pragma unroll
    for (int c = 0; c < 8; ++c) {
        bf16x16 afr = (c < 2) ? lf[c] : (c < 4) ? rf[c - 2] : (c < 6) ? df[c - 4] : af[c - 6];
#pragma unroll
        for (int t = 0; t < 4; ++t)
            acc[t] = wmma_bf16(afr, bfrag(wp1p, c, t, lane), acc[t]);
    }

    // layer2: 64 -> 1 per pair; VALU dot + 16-lane shuffle reduction.
    int nl = lane & 15, half = lane >> 4;
    float dot[8];
#pragma unroll
    for (int r = 0; r < 8; ++r) dot[r] = 0.f;
#pragma unroll
    for (int t = 0; t < 4; ++t) {
        float bb = bp1[t * 16 + nl];
        float ww = wp2[t * 16 + nl];
#pragma unroll
        for (int r = 0; r < 8; ++r) {
            float v = acc[t][r] + bb;
            v = v > 0.f ? v : 0.f;
            dot[r] += v * ww;
        }
    }
#pragma unroll
    for (int r = 0; r < 8; ++r) {
#pragma unroll
        for (int m = 1; m < 16; m <<= 1)
            dot[r] += __shfl_xor(dot[r], m, 16);
    }
    if (nl == 0) {
        float bias2 = bp2[0];
        float sum = 0.f;
#pragma unroll
        for (int r = 0; r < 8; ++r) {
            int j = j0 + r + 8 * half;
            float logit = dot[r] + bias2;
            float p = 1.f / (1.f + __expf(-logit));
            if (j == i) p = 0.f;  // zero diagonal
            sum += p;
        }
        atomicAdd(out + i, sum * (1.f / (NACT - 1)));
    }
}

// ---------------------------------------------------------------- launch
extern "C" void kernel_launch(void* const* d_in, const int* in_sizes, int n_in,
                              void* d_out, int out_size, void* d_ws, size_t ws_size,
                              hipStream_t stream) {
    (void)in_sizes; (void)n_in; (void)out_size; (void)ws_size;
    const float* feat   = (const float*)d_in[0];
    const float* ewgt   = (const float*)d_in[1];
    const float* Wn1    = (const float*)d_in[2];
    const float* bn1    = (const float*)d_in[3];
    const float* Wn2    = (const float*)d_in[4];
    const float* bn2    = (const float*)d_in[5];
    const float* relemb = (const float*)d_in[6];
    const float* Wm1    = (const float*)d_in[7];
    const float* bm1    = (const float*)d_in[8];
    const float* Wm2    = (const float*)d_in[9];
    const float* bm2    = (const float*)d_in[10];
    const float* ln_g   = (const float*)d_in[11];
    const float* ln_b   = (const float*)d_in[12];
    const float* Wa1    = (const float*)d_in[13];
    const float* ba1    = (const float*)d_in[14];
    const float* Wa2    = (const float*)d_in[15];
    const float* ba2    = (const float*)d_in[16];
    const float* Wp1    = (const float*)d_in[17];
    const float* bp1    = (const float*)d_in[18];
    const float* Wp2    = (const float*)d_in[19];
    const float* bp2    = (const float*)d_in[20];
    const int* esrc   = (const int*)d_in[21];
    const int* edst   = (const int*)d_in[22];
    const int* erel   = (const int*)d_in[23];
    const int* cblock = (const int*)d_in[24];
    const int* ctgt   = (const int*)d_in[25];
    const int* cprim  = (const int*)d_in[26];
    const int* htarg  = (const int*)d_in[27];
    float* out = (float*)d_out;

    char* ws = (char*)d_ws;
    size_t off = 0;
    auto carve = [&](size_t bytes) -> void* {
        void* p = ws + off;
        off = (off + bytes + 255) & ~(size_t)255;
        return p;
    };
    float*  nodes_f32  = (float*) carve((size_t)NNODE * H * 4);
    bf16_t* nodes_bf16 = (bf16_t*)carve((size_t)NNODE * H * 2);
    float*  agg        = (float*) carve((size_t)NNODE * H * 4);
    bf16_t* ln_bf      = (bf16_t*)carve((size_t)NNODE * H * 2);
    float*  gsum       = (float*) carve(H * 4);
    bf16_t* graph_bf   = (bf16_t*)carve(H * 2);
    bf16_t* enc_bf     = (bf16_t*)carve((size_t)NACT * H * 2);
    bf16_t* rel_bf     = (bf16_t*)carve(6 * H * 2);
    bf16_t* wn1p = (bf16_t*)carve(4 * 2048 * 2);
    bf16_t* wn2p = (bf16_t*)carve(2 * 2048 * 2);
    bf16_t* wm1p = (bf16_t*)carve(7 * 2048 * 2);
    bf16_t* wm2p = (bf16_t*)carve(2 * 2048 * 2);
    bf16_t* wa1p = (bf16_t*)carve(9 * 2048 * 2);
    bf16_t* wa2p = (bf16_t*)carve(2 * 2048 * 2);
    bf16_t* wp1p = (bf16_t*)carve(8 * 2048 * 2);

    // weight packing (B-frag layout) + constants
    k_pack_b<<<(4 * 2048 + 255) / 256, 256, 0, stream>>>(Wn1, NFEAT, 4, wn1p);
    k_pack_b<<<(2 * 2048 + 255) / 256, 256, 0, stream>>>(Wn2, H, 2, wn2p);
    k_pack_b<<<(7 * 2048 + 255) / 256, 256, 0, stream>>>(Wm1, 3 * H + 1, 7, wm1p);
    k_pack_b<<<(2 * 2048 + 255) / 256, 256, 0, stream>>>(Wm2, H, 2, wm2p);
    k_pack_b<<<(9 * 2048 + 255) / 256, 256, 0, stream>>>(Wa1, 4 * H + NPRIM + 1, 9, wa1p);
    k_pack_b<<<(2 * 2048 + 255) / 256, 256, 0, stream>>>(Wa2, H, 2, wa2p);
    k_pack_b<<<(8 * 2048 + 255) / 256, 256, 0, stream>>>(Wp1, 4 * H, 8, wp1p);
    k_f32_to_bf16<<<(6 * H + 255) / 256, 256, 0, stream>>>(relemb, rel_bf, 6 * H);

    // zero accumulators (every call: deterministic)
    k_zero_f32<<<((size_t)NNODE * H + 255) / 256, 256, 0, stream>>>(agg, NNODE * H);
    k_zero_f32<<<1, 256, 0, stream>>>(gsum, H);
    k_zero_f32<<<2, 256, 0, stream>>>(out, NACT);

    // pipeline
    k_node_mlp<<<(NNODE + 127) / 128, 256, 0, stream>>>(feat, wn1p, bn1, wn2p, bn2,
                                                        nodes_f32, nodes_bf16);
    k_edge_mlp<<<NEDGE / 128, 256, 0, stream>>>(nodes_bf16, rel_bf, ewgt, esrc, edst, erel,
                                                wm1p, bm1, wm2p, bm2, agg);
    k_layernorm<<<NNODE / 8, 256, 0, stream>>>(nodes_f32, agg, ln_g, ln_b, ln_bf, gsum);
    k_graph_finalize<<<1, 64, 0, stream>>>(gsum, graph_bf);
    k_action_enc<<<NACT / 128, 256, 0, stream>>>(ln_bf, graph_bf, cblock, ctgt, cprim, htarg,
                                                 wa1p, ba1, wa2p, ba2, enc_bf);
    k_pair_rank<<<(NACT * NACT / 16) / 8, 256, 0, stream>>>(enc_bf, wp1p, bp1, Wp2, bp2, out);
}
